// RoPEMultiheadAttention_5093831213170
// MI455X (gfx1250) — matrix operand order repaired and checked by reading, not compiled, not run on hardware
//
#include <hip/hip_runtime.h>
#include <hip/hip_bf16.h>

// ---------------------------------------------------------------------------
// RoPE Multi-Head Attention for MI455X (gfx1250), wave32 + WMMA f16 +
// async global->LDS staging (ASYNCcnt) for the attention K/V stream.
// B=2, S=2048, D=1024, H=16, Dh=64.
// ---------------------------------------------------------------------------

#define B_     2
#define S_     2048
#define D_     1024
#define H_     16
#define DH_    64
#define M_     (B_ * S_)        // 4096 flattened rows

typedef __attribute__((ext_vector_type(16))) _Float16 v16h;
typedef __attribute__((ext_vector_type(8)))  _Float16 v8h;
typedef __attribute__((ext_vector_type(4)))  _Float16 v4h;
typedef __attribute__((ext_vector_type(8)))  float    v8f;
typedef __attribute__((ext_vector_type(4)))  float    v4f;

union Frag { v16h v; v8h h[2]; };

__device__ __forceinline__ v8f wmma_f16(const v16h& a, const v16h& b, const v8f& c) {
    return __builtin_amdgcn_wmma_f32_16x16x32_f16(false, a, false, b, (short)0, c, false, false);
}

// ---------------------------------------------------------------------------
// Kernel 0: fp32 -> f16 conversion (vectorized x4)
// ---------------------------------------------------------------------------
__global__ void cvt_f32_f16(const float* __restrict__ src, _Float16* __restrict__ dst, int n4) {
    int i = blockIdx.x * blockDim.x + threadIdx.x;
    int stride = gridDim.x * blockDim.x;
    for (; i < n4; i += stride) {
        v4f s = *(const v4f*)(src + 4 * (size_t)i);
        v4h d;
        d.x = (_Float16)s.x; d.y = (_Float16)s.y;
        d.z = (_Float16)s.z; d.w = (_Float16)s.w;
        *(v4h*)(dst + 4 * (size_t)i) = d;
    }
}

// ---------------------------------------------------------------------------
// Kernel 1: projection GEMM Y = X @ W^T (+ RoPE for q/k, transposed store for v)
// Software-pipelined K loop: load frags for k+1 before issuing WMMAs for k.
// mode 0: Q (RoPE, row-major)  mode 1: K (RoPE, row-major)  mode 2: V -> Vt[b][h][d][s]
// ---------------------------------------------------------------------------
__global__ void proj_rope_kernel(const _Float16* __restrict__ X,
                                 const _Float16* __restrict__ W,
                                 _Float16* __restrict__ Y, int mode) {
    const int wave = threadIdx.x >> 5;
    const int lane = threadIdx.x & 31;
    const int half = lane >> 4;
    const int l16  = lane & 15;

    const int strip = blockIdx.x * 4 + wave;     // 4096 strips total
    const int m0 = (strip >> 4) << 4;            // 256 M-strips of 16
    const int n0 = (strip & 15) << 6;            // 16 N-strips of 64 (== head)

    const _Float16* abase  = X + (size_t)(m0 + l16) * D_ + (half ? 16 : 0);
    const _Float16* bbase0 = W + (size_t)(n0 + l16) * D_ + (half ? 16 : 0);

    auto ldA = [&](Frag& f, int kk) {
        f.h[0] = *(const v8h*)(abase + kk);
        f.h[1] = *(const v8h*)(abase + kk + 8);
    };
    auto ldB = [&](Frag& f, int kk, int nt) {
        const _Float16* bb = bbase0 + (size_t)nt * 16 * D_ + kk;
        f.h[0] = *(const v8h*)(bb);
        f.h[1] = *(const v8h*)(bb + 8);
    };

    v8f acc[4] = {};
    Frag ac, bc[4];
    ldA(ac, 0);
#pragma unroll
    for (int nt = 0; nt < 4; ++nt) ldB(bc[nt], 0, nt);

    for (int kk = 0; kk < D_ - 32; kk += 32) {
        Frag an, bn[4];
        ldA(an, kk + 32);
#pragma unroll
        for (int nt = 0; nt < 4; ++nt) ldB(bn[nt], kk + 32, nt);
        __builtin_prefetch(abase + kk + 96, 0, 1);
#pragma unroll
        for (int nt = 0; nt < 4; ++nt) acc[nt] = wmma_f16(ac.v, bc[nt].v, acc[nt]);
        ac = an;
#pragma unroll
        for (int nt = 0; nt < 4; ++nt) bc[nt] = bn[nt];
    }
#pragma unroll
    for (int nt = 0; nt < 4; ++nt) acc[nt] = wmma_f16(ac.v, bc[nt].v, acc[nt]);

    // ---- RoPE (modes 0,1). Pair (d, d+32) lives in (acc[nt], acc[nt+2]). ----
    if (mode < 2) {
#pragma unroll
        for (int nt = 0; nt < 2; ++nt) {
            const int d = nt * 16 + l16;                      // 0..31 pair index
            const float invf = __powf(10000.0f, -(float)d * (1.0f / 32.0f));
#pragma unroll
            for (int j = 0; j < 8; ++j) {
                const int m   = m0 + j + 8 * half;
                const int pos = m & (S_ - 1);
                const float f = (float)pos * invf;
                const float c = __cosf(f), s = __sinf(f);
                const float x1 = acc[nt][j], x2 = acc[nt + 2][j];
                acc[nt][j]     = x1 * c - x2 * s;
                acc[nt + 2][j] = x1 * s + x2 * c;
            }
        }
    }

    // ---- store ----
    if (mode < 2) {                       // row-major (m, n) f16
#pragma unroll
        for (int nt = 0; nt < 4; ++nt) {
            const int col = n0 + nt * 16 + l16;
#pragma unroll
            for (int j = 0; j < 8; ++j) {
                const int m = m0 + j + 8 * half;
                Y[(size_t)m * D_ + col] = (_Float16)acc[nt][j];
            }
        }
    } else {                              // Vt[b][h][d][s]
        const int h = n0 >> 6;
#pragma unroll
        for (int nt = 0; nt < 4; ++nt) {
            const int d = nt * 16 + l16;
#pragma unroll
            for (int j = 0; j < 8; ++j) {
                const int m = m0 + j + 8 * half;
                const int b = m >> 11, s = m & (S_ - 1);
                Y[(((size_t)(b * H_ + h) * DH_) + d) * S_ + s] = (_Float16)acc[nt][j];
            }
        }
    }
}

// ---------------------------------------------------------------------------
// Kernel 2: attention. block = 128 (4 waves, 64 q-rows of one (b,h)).
// K/V chunks (64 kv rows) are DMAed once per block into double-buffered LDS
// via GLOBAL_LOAD_ASYNC_TO_LDS_B128 (ASYNCcnt), overlapping copy(ch+1) with
// compute(ch). LDS rows padded to 72 halves for bank-conflict-free ds_read_b128.
// Pass 1: online row max/sum (64-wide updates).
// Pass 2: recompute, normalize, stream P (537MB written exactly once) + P·V.
// ---------------------------------------------------------------------------
__global__ void attn_kernel(const _Float16* __restrict__ Qr,
                            const _Float16* __restrict__ Kr,
                            const _Float16* __restrict__ Vt,
                            float* __restrict__ attn_out,
                            _Float16* __restrict__ Ctx) {
    const int tid  = threadIdx.x;
    const int wave = tid >> 5;
    const int lane = tid & 31;
    const int half = lane >> 4;
    const int l16  = lane & 15;

    const int bh = blockIdx.y;
    const int b  = bh >> 4, h = bh & 15;
    const int qm0 = blockIdx.x * 64 + wave * 16;
    const int gq  = b * S_ + qm0;                 // global flattened q row base
    const float scale = 0.125f;                   // 1/sqrt(64)

    __shared__ __align__(16) _Float16 kl[2][64][72];   // K chunk, padded rows
    __shared__ __align__(16) _Float16 vl[2][64][72];   // V chunk, padded rows
    __shared__ __align__(16) _Float16 pl[4][16][32];   // per-wave P staging

    // Q A-fragments (registers) for the two d-steps (d 0..31, 32..63).
    const _Float16* qbase = Qr + (size_t)(gq + l16) * D_ + h * DH_ + (half ? 16 : 0);
    Frag qa0, qa1;
    qa0.h[0] = *(const v8h*)(qbase);
    qa0.h[1] = *(const v8h*)(qbase + 8);
    qa1.h[0] = *(const v8h*)(qbase + 32);
    qa1.h[1] = *(const v8h*)(qbase + 40);

    const _Float16* Kh = Kr + (size_t)(b * S_) * D_ + h * DH_;   // [s][64]
    const _Float16* Vh = Vt + (size_t)bh * DH_ * S_;             // [d][s]

    // async DMA of one 64-row chunk (64 x 128B) into LDS: 512 x 16B pieces,
    // 4 per thread.
    auto copyK = [&](int ch, int buf) {
#pragma unroll
        for (int i = 0; i < 4; ++i) {
            const int c = tid + i * 128;
            const int row = c >> 3, seg = c & 7;
            const _Float16* g = Kh + (size_t)(ch * 64 + row) * D_ + seg * 8;
            unsigned l = (unsigned)(size_t)&kl[buf][row][seg * 8];
            asm volatile("global_load_async_to_lds_b128 %0, %1, off"
                         :: "v"(l), "v"(g) : "memory");
        }
    };
    auto copyV = [&](int ch, int buf) {
#pragma unroll
        for (int i = 0; i < 4; ++i) {
            const int c = tid + i * 128;
            const int d = c >> 3, seg = c & 7;
            const _Float16* g = Vh + (size_t)d * S_ + ch * 64 + seg * 8;
            unsigned l = (unsigned)(size_t)&vl[buf][d][seg * 8];
            asm volatile("global_load_async_to_lds_b128 %0, %1, off"
                         :: "v"(l), "v"(g) : "memory");
        }
    };

    float rmax[8], rsum[8];
#pragma unroll
    for (int j = 0; j < 8; ++j) { rmax[j] = -1e30f; rsum[j] = 0.0f; }

    // ---------------- pass 1: row stats (64 kv per step) ----------------
    copyK(0, 0);
    asm volatile("s_wait_asynccnt 0" ::: "memory");
    __syncthreads();
    for (int ch = 0; ch < S_ / 64; ++ch) {
        const int buf = ch & 1;
        if (ch + 1 < S_ / 64) copyK(ch + 1, buf ^ 1);   // overlap with compute

        v8f sc[4];
#pragma unroll
        for (int t = 0; t < 4; ++t) {
            const _Float16* kp = &kl[buf][t * 16 + l16][half ? 16 : 0];
            Frag k0, k1;
            k0.h[0] = *(const v8h*)(kp);      k0.h[1] = *(const v8h*)(kp + 8);
            k1.h[0] = *(const v8h*)(kp + 32); k1.h[1] = *(const v8h*)(kp + 40);
            v8f s_ = {};
            s_ = wmma_f16(qa0.v, k0.v, s_);
            s_ = wmma_f16(qa1.v, k1.v, s_);
            sc[t] = s_;
        }
#pragma unroll
        for (int j = 0; j < 8; ++j) {
            const float s0 = sc[0][j] * scale, s1 = sc[1][j] * scale;
            const float s2 = sc[2][j] * scale, s3 = sc[3][j] * scale;
            float tm = fmaxf(fmaxf(s0, s1), fmaxf(s2, s3));
            tm = fmaxf(tm, __shfl_xor(tm, 1));
            tm = fmaxf(tm, __shfl_xor(tm, 2));
            tm = fmaxf(tm, __shfl_xor(tm, 4));
            tm = fmaxf(tm, __shfl_xor(tm, 8));
            const float nm = fmaxf(rmax[j], tm);
            float e = __expf(s0 - nm) + __expf(s1 - nm) +
                      __expf(s2 - nm) + __expf(s3 - nm);
            e += __shfl_xor(e, 1);
            e += __shfl_xor(e, 2);
            e += __shfl_xor(e, 4);
            e += __shfl_xor(e, 8);
            rsum[j] = rsum[j] * __expf(rmax[j] - nm) + e;
            rmax[j] = nm;
        }
        asm volatile("s_wait_asynccnt 0" ::: "memory");
        __syncthreads();
    }
#pragma unroll
    for (int j = 0; j < 8; ++j) rsum[j] = 1.0f / rsum[j];

    // ---------------- pass 2: P out + P·V ----------------
    copyK(0, 0); copyV(0, 0);
    asm volatile("s_wait_asynccnt 0" ::: "memory");
    __syncthreads();

    float* aout = attn_out + ((size_t)bh * S_ + qm0) * S_;
    v8f cacc[4] = {};

    for (int ch = 0; ch < S_ / 64; ++ch) {
        const int buf = ch & 1;
        if (ch + 1 < S_ / 64) { copyK(ch + 1, buf ^ 1); copyV(ch + 1, buf ^ 1); }

#pragma unroll
        for (int kt2l = 0; kt2l < 2; ++kt2l) {            // 32 kv per sub-step
#pragma unroll
            for (int sub = 0; sub < 2; ++sub) {
                const int tt = kt2l * 2 + sub;            // 16-tile within chunk
                const int kt = ch * 4 + tt;               // global 16-tile index
                const _Float16* kp = &kl[buf][tt * 16 + l16][half ? 16 : 0];
                Frag k0, k1;
                k0.h[0] = *(const v8h*)(kp);      k0.h[1] = *(const v8h*)(kp + 8);
                k1.h[0] = *(const v8h*)(kp + 32); k1.h[1] = *(const v8h*)(kp + 40);
                v8f sc = {};
                sc = wmma_f16(qa0.v, k0.v, sc);
                sc = wmma_f16(qa1.v, k1.v, sc);
#pragma unroll
                for (int j = 0; j < 8; ++j) {
                    const int row = j + 8 * half;
                    const float p = __expf(sc[j] * scale - rmax[j]) * rsum[j];
                    aout[(size_t)row * S_ + kt * 16 + l16] = p;   // streamed once
                    pl[wave][row][sub * 16 + l16] = (_Float16)p;
                }
            }
            // cross-lane LDS store->load hazard within the wave
            asm volatile("s_wait_dscnt 0" ::: "memory");

            Frag pa;
            const _Float16* pb = &pl[wave][l16][half ? 16 : 0];
            pa.h[0] = *(const v8h*)(pb);
            pa.h[1] = *(const v8h*)(pb + 8);

#pragma unroll
            for (int dt = 0; dt < 4; ++dt) {
                const _Float16* vp = &vl[buf][dt * 16 + l16][kt2l * 32 + (half ? 16 : 0)];
                Frag vb;
                vb.h[0] = *(const v8h*)(vp);
                vb.h[1] = *(const v8h*)(vp + 8);
                cacc[dt] = wmma_f16(pa.v, vb.v, cacc[dt]);
            }
        }
        asm volatile("s_wait_asynccnt 0" ::: "memory");
        __syncthreads();
    }

    // ctx -> row-major (m, h*64+d) f16
#pragma unroll
    for (int dt = 0; dt < 4; ++dt) {
        const int col = h * DH_ + dt * 16 + l16;
#pragma unroll
        for (int j = 0; j < 8; ++j) {
            const int m = gq + j + 8 * half;
            Ctx[(size_t)m * D_ + col] = (_Float16)cacc[dt][j];
        }
    }
}

// ---------------------------------------------------------------------------
// Kernel 3: output projection  out = Ctx @ Wo^T + bo  (fp32 out), pipelined.
// ---------------------------------------------------------------------------
__global__ void out_proj_kernel(const _Float16* __restrict__ X,
                                const _Float16* __restrict__ W,
                                const float* __restrict__ bo,
                                float* __restrict__ out) {
    const int wave = threadIdx.x >> 5;
    const int lane = threadIdx.x & 31;
    const int half = lane >> 4;
    const int l16  = lane & 15;

    const int strip = blockIdx.x * 4 + wave;
    const int m0 = (strip >> 4) << 4;
    const int n0 = (strip & 15) << 6;

    const _Float16* abase  = X + (size_t)(m0 + l16) * D_ + (half ? 16 : 0);
    const _Float16* bbase0 = W + (size_t)(n0 + l16) * D_ + (half ? 16 : 0);

    auto ldA = [&](Frag& f, int kk) {
        f.h[0] = *(const v8h*)(abase + kk);
        f.h[1] = *(const v8h*)(abase + kk + 8);
    };
    auto ldB = [&](Frag& f, int kk, int nt) {
        const _Float16* bb = bbase0 + (size_t)nt * 16 * D_ + kk;
        f.h[0] = *(const v8h*)(bb);
        f.h[1] = *(const v8h*)(bb + 8);
    };

    v8f acc[4] = {};
    Frag ac, bc[4];
    ldA(ac, 0);
#pragma unroll
    for (int nt = 0; nt < 4; ++nt) ldB(bc[nt], 0, nt);

    for (int kk = 0; kk < D_ - 32; kk += 32) {
        Frag an, bn[4];
        ldA(an, kk + 32);
#pragma unroll
        for (int nt = 0; nt < 4; ++nt) ldB(bn[nt], kk + 32, nt);
        __builtin_prefetch(abase + kk + 96, 0, 1);
#pragma unroll
        for (int nt = 0; nt < 4; ++nt) acc[nt] = wmma_f16(ac.v, bc[nt].v, acc[nt]);
        ac = an;
#pragma unroll
        for (int nt = 0; nt < 4; ++nt) bc[nt] = bn[nt];
    }
#pragma unroll
    for (int nt = 0; nt < 4; ++nt) acc[nt] = wmma_f16(ac.v, bc[nt].v, acc[nt]);

#pragma unroll
    for (int nt = 0; nt < 4; ++nt) {
        const int col = n0 + nt * 16 + l16;
        const float bias = bo[col];
#pragma unroll
        for (int j = 0; j < 8; ++j) {
            const int m = m0 + j + 8 * half;
            out[(size_t)m * D_ + col] = acc[nt][j] + bias;
        }
    }
}

// ---------------------------------------------------------------------------
// Host launcher
// ---------------------------------------------------------------------------
extern "C" void kernel_launch(void* const* d_in, const int* in_sizes, int n_in,
                              void* d_out, int out_size, void* d_ws, size_t ws_size,
                              hipStream_t stream) {
    const float* q_in  = (const float*)d_in[0];
    const float* k_in  = (const float*)d_in[1];
    const float* v_in  = (const float*)d_in[2];
    const float* Wq    = (const float*)d_in[3];
    const float* Wk    = (const float*)d_in[4];
    const float* Wv    = (const float*)d_in[5];
    const float* Wo    = (const float*)d_in[6];
    const float* bo    = (const float*)d_in[7];

    float* out      = (float*)d_out;                          // (B,S,D) fp32
    float* attn_out = (float*)d_out + (size_t)M_ * D_;        // (B,H,S,S) fp32

    // workspace carve-up (f16), 64 MB total
    _Float16* w = (_Float16*)d_ws;
    size_t o = 0;
    _Float16* Xq  = w + o; o += (size_t)M_ * D_;
    _Float16* Xk  = w + o; o += (size_t)M_ * D_;
    _Float16* Xv  = w + o; o += (size_t)M_ * D_;
    _Float16* Wq6 = w + o; o += (size_t)D_ * D_;
    _Float16* Wk6 = w + o; o += (size_t)D_ * D_;
    _Float16* Wv6 = w + o; o += (size_t)D_ * D_;
    _Float16* Wo6 = w + o; o += (size_t)D_ * D_;
    _Float16* Qr  = w + o; o += (size_t)M_ * D_;
    _Float16* Kr  = w + o; o += (size_t)M_ * D_;
    _Float16* Vt  = w + o; o += (size_t)M_ * D_;
    _Float16* Ctx = w + o; o += (size_t)M_ * D_;
    (void)ws_size; (void)in_sizes; (void)n_in; (void)out_size;

    // 0) convert inputs to f16
    const int act4 = (M_ * D_) / 4, wt4 = (D_ * D_) / 4;
    cvt_f32_f16<<<1024, 256, 0, stream>>>(q_in, Xq, act4);
    cvt_f32_f16<<<1024, 256, 0, stream>>>(k_in, Xk, act4);
    cvt_f32_f16<<<1024, 256, 0, stream>>>(v_in, Xv, act4);
    cvt_f32_f16<<<512,  256, 0, stream>>>(Wq, Wq6, wt4);
    cvt_f32_f16<<<512,  256, 0, stream>>>(Wk, Wk6, wt4);
    cvt_f32_f16<<<512,  256, 0, stream>>>(Wv, Wv6, wt4);
    cvt_f32_f16<<<512,  256, 0, stream>>>(Wo, Wo6, wt4);

    // 1) projections (+RoPE for q,k; transposed store for v)
    const int proj_blocks = (M_ / 16) * (D_ / 64) / 4;        // 1024
    proj_rope_kernel<<<proj_blocks, 128, 0, stream>>>(Xq, Wq6, Qr, 0);
    proj_rope_kernel<<<proj_blocks, 128, 0, stream>>>(Xk, Wk6, Kr, 1);
    proj_rope_kernel<<<proj_blocks, 128, 0, stream>>>(Xv, Wv6, Vt, 2);

    // 2) attention: grid (S/64 q-blocks, B*H)
    attn_kernel<<<dim3(S_ / 64, B_ * H_), 128, 0, stream>>>(Qr, Kr, Vt, attn_out, Ctx);

    // 3) output projection + bias
    out_proj_kernel<<<proj_blocks, 128, 0, stream>>>(Ctx, Wo6, bo, out);
}
